// VertexHybridModel_51479478010132
// MI455X (gfx1250) — matrix operand order repaired and checked by reading, not compiled, not run on hardware
//
#include <hip/hip_runtime.h>
#include <hip/hip_bf16.h>

typedef __attribute__((ext_vector_type(16))) _Float16 v16h;
typedef __attribute__((ext_vector_type(8)))  _Float16 v8h;
typedef __attribute__((ext_vector_type(8)))  float    v8f;

// ---------------------------------------------------------------------------
// WMMA fragment layout (gfx1250 wave32, V_WMMA_F32_16X16X32_F16)
// A 16x32 f16: lane<16 row=lane, K {0..7,16..23}; lane>=16 row=lane-16, K {8..15,24..31}
// B 32x16 f16: lane<16 col=lane, K {0..7,16..23}; lane>=16 col=lane-16, K {8..15,24..31}
// C/D 16x16 f32: vgpr i -> M = i + (lane>=16 ? 8:0), N = lane&15
// ---------------------------------------------------------------------------

__device__ inline float warp_max32(float v) {
#pragma unroll
  for (int o = 16; o > 0; o >>= 1) v = fmaxf(v, __shfl_xor(v, o, 32));
  return v;
}
__device__ inline float warp_sum32(float v) {
#pragma unroll
  for (int o = 16; o > 0; o >>= 1) v += __shfl_xor(v, o, 32);
  return v;
}

__device__ inline v16h cat8(v8h lo, v8h hi) {
  return __builtin_shufflevector(lo, hi, 0, 1, 2, 3, 4, 5, 6, 7,
                                 8, 9, 10, 11, 12, 13, 14, 15);
}

// ---------------------------------------------------------------------------
// Linear: out[M,Nc] = act(X[M,K] @ W[Nc,K]^T + bias) + add0 + add1
// One wave -> 16 rows x (NT*16) cols strip; A fragment reused across NT WMMAs.
// K multiple of 32, M multiple of 16. f32 operands, cvt on the fly (table k order, two runs of 8).
// ---------------------------------------------------------------------------
template <int NT>
__global__ void k_linear_t(const float* __restrict__ X, const float* __restrict__ W,
                           const float* __restrict__ bias,
                           const float* __restrict__ add0, const float* __restrict__ add1,
                           float* __restrict__ out, int M, int K, int Nc, int do_relu) {
  const int lane = threadIdx.x & 31;
  const int strip = NT * 16;
  const int tnc = (Nc + strip - 1) / strip;
  const int total = (M >> 4) * tnc;
  int wg = blockIdx.x * (blockDim.x >> 5) + (threadIdx.x >> 5);
  if (wg >= total) return;
  const int tm = wg / tnc, tn = wg % tnc;
  const int m0 = tm << 4, n0 = tn * strip;
  const int rowA = m0 + (lane & 15);
  const int abase = (lane & 16) ? 8 : 0;
  const int bbase = (lane & 16) ? 8 : 0;
  int col[NT];
  bool bvalid[NT];
#pragma unroll
  for (int t = 0; t < NT; ++t) {
    col[t] = n0 + 16 * t + (lane & 15);
    bvalid[t] = col[t] < Nc;
  }
  const v8f vzero = {};
  v8f acc[NT];
#pragma unroll
  for (int t = 0; t < NT; ++t) acc[t] = vzero;

  for (int k0 = 0; k0 < K; k0 += 32) {
    v16h a;
    const float* pa = X + (size_t)rowA * K + k0 + abase;
#pragma unroll
    for (int e = 0; e < 8; ++e)  a[e] = (_Float16)pa[e];
#pragma unroll
    for (int e = 8; e < 16; ++e) a[e] = (_Float16)pa[e + 8];
    if (k0 + 32 < K) __builtin_prefetch(pa + 32, 0, 3);
#pragma unroll
    for (int t = 0; t < NT; ++t) {
      v16h b;
      if (bvalid[t]) {
        const float* pb = W + (size_t)col[t] * K + k0 + bbase;
#pragma unroll
        for (int e = 0; e < 16; ++e) b[e] = (_Float16)pb[e < 8 ? e : e + 8];
      } else {
#pragma unroll
        for (int e = 0; e < 16; ++e) b[e] = (_Float16)0.0f;
      }
      acc[t] = __builtin_amdgcn_wmma_f32_16x16x32_f16(false, a, false, b, (short)0,
                                                      acc[t], false, false);
    }
  }

  const int mofs = (lane & 16) ? 8 : 0;
#pragma unroll
  for (int t = 0; t < NT; ++t) {
    if (!bvalid[t]) continue;
    const float bs = bias ? bias[col[t]] : 0.0f;
#pragma unroll
    for (int i = 0; i < 8; ++i) {
      const int m = m0 + mofs + i;
      float v = acc[t][i] + bs;
      if (do_relu) v = fmaxf(v, 0.0f);
      if (add0) v += add0[(size_t)m * Nc + col[t]];
      if (add1) v += add1[(size_t)m * Nc + col[t]];
      out[(size_t)m * Nc + col[t]] = v;
    }
  }
}

// ---------------------------------------------------------------------------
// Batched f16 GEMM: out[b] = P16[b](Nn x Nn) @ XT[b]^T  [+ (1+eps)*Xadd[b]]
//   P16: f16 row-major (Nn x Nn); XT: f16 (Hc x Nn) = B operand pre-transposed
//   so every fragment is two contiguous 16B loads. One wave -> 16x64 strip.
// ---------------------------------------------------------------------------
__global__ void k_bmm16(const _Float16* __restrict__ P16, const _Float16* __restrict__ XT,
                        const float* __restrict__ Xadd, const float* __restrict__ epsp,
                        float* __restrict__ out, int Bc, int Nn, int Hc) {
  const int lane = threadIdx.x & 31;
  const int tmc = Nn >> 4, tnc = Hc >> 6;  // strips of 64 cols
  const int per_b = tmc * tnc;
  int wg = blockIdx.x * (blockDim.x >> 5) + (threadIdx.x >> 5);
  if (wg >= Bc * per_b) return;
  const int b = wg / per_b;
  const int t0 = wg % per_b;
  const int m0 = (t0 / tnc) << 4, n0 = (t0 % tnc) << 6;
  const _Float16* Pb = P16 + (size_t)b * Nn * Nn;
  const _Float16* XTb = XT + (size_t)b * Hc * Nn;
  float* Ob = out + (size_t)b * Nn * Hc;
  const int rowA = m0 + (lane & 15);
  const int abase = (lane & 16) ? 8 : 0;
  const int bbase = (lane & 16) ? 8 : 0;
  int col[4];
#pragma unroll
  for (int t = 0; t < 4; ++t) col[t] = n0 + 16 * t + (lane & 15);
  const v8f vzero = {};
  v8f acc[4];
#pragma unroll
  for (int t = 0; t < 4; ++t) acc[t] = vzero;

  const _Float16* parow = Pb + (size_t)rowA * Nn;
  for (int k0 = 0; k0 < Nn; k0 += 32) {
    const v8h* pa = (const v8h*)(parow + k0 + abase);
    const v16h a = cat8(pa[0], pa[2]);  // K {0..7} and {16..23} (half-wave shifted)
    if (k0 + 32 < Nn) __builtin_prefetch(parow + k0 + 32, 0, 3);
#pragma unroll
    for (int t = 0; t < 4; ++t) {
      const v8h* pb = (const v8h*)(XTb + (size_t)col[t] * Nn + k0 + bbase);
      const v16h bf = cat8(pb[0], pb[2]);  // K {0..7} and {16..23} (half-wave shifted)
      acc[t] = __builtin_amdgcn_wmma_f32_16x16x32_f16(false, a, false, bf, (short)0,
                                                      acc[t], false, false);
    }
  }

  const int mofs = (lane & 16) ? 8 : 0;
  const float s = epsp ? (1.0f + epsp[0]) : 0.0f;
#pragma unroll
  for (int t = 0; t < 4; ++t) {
#pragma unroll
    for (int i = 0; i < 8; ++i) {
      const int m = m0 + mofs + i;
      float v = acc[t][i];
      if (epsp) v += s * Xadd[(size_t)b * Nn * Hc + (size_t)m * Hc + col[t]];
      Ob[(size_t)m * Hc + col[t]] = v;
    }
  }
}

// ---------------------------------------------------------------------------
// f32 -> f16 convert (adjacency matrices, exact for {0,1})
// ---------------------------------------------------------------------------
__global__ void k_cvt_f16(const float* __restrict__ in, _Float16* __restrict__ o, int n) {
  const int i = blockIdx.x * 256 + threadIdx.x;
  if (i < n) o[i] = (_Float16)in[i];
}

// ---------------------------------------------------------------------------
// Tiled transpose+convert: X (B,Nn,Hc) f32 -> XT (B,Hc,Nn) f16. 32x33 LDS tile.
// ---------------------------------------------------------------------------
__global__ void k_transpose16(const float* __restrict__ X, _Float16* __restrict__ XT,
                              int Nn, int Hc) {
  __shared__ float tile[32][33];
  const int b = blockIdx.z;
  const int n0 = blockIdx.x * 32, h0 = blockIdx.y * 32;
  const int c = threadIdx.x & 31, r8 = threadIdx.x >> 5;  // 256 threads
  const float* Xb = X + ((size_t)b * Nn + n0) * Hc + h0;
#pragma unroll
  for (int rr = 0; rr < 32; rr += 8)
    tile[rr + r8][c] = Xb[(size_t)(rr + r8) * Hc + c];
  __syncthreads();
  _Float16* XTb = XT + ((size_t)b * Hc + h0) * Nn + n0;
#pragma unroll
  for (int rr = 0; rr < 32; rr += 8)
    XTb[(size_t)(rr + r8) * Nn + c] = (_Float16)tile[c][rr + r8];
}

// ---------------------------------------------------------------------------
// Flexible attention row kernel: one block per (b,i). alpha written as f16.
// Deterministic: shuffle-tree reductions + fixed-order partial sums.
// ---------------------------------------------------------------------------
__global__ void k_attn(const float* __restrict__ HO, const float* __restrict__ H1,
                       const float* __restrict__ wphi, _Float16* __restrict__ alpha,
                       float* __restrict__ ctx) {
  const int ATT = 32, NN = 512;
  const int bi = blockIdx.x;        // b*N + i
  const int b = bi >> 9;
  const int tid = threadIdx.x;      // 256 threads = 8 waves
  const int lane = tid & 31, wid = tid >> 5;
  __shared__ float s_ho[32], s_phi[32], s_s[512], s_red[8], s_cp[8 * 32];
  if (tid < ATT) { s_ho[tid] = HO[(size_t)bi * ATT + tid]; s_phi[tid] = wphi[tid]; }
  __syncthreads();

  // pass 1: scores + max
  float lmax = -3.0e38f;
  for (int j = tid; j < NN; j += 256) {
    const float* hj = H1 + (size_t)(b * NN + j) * ATT;
    float s = 0.0f;
#pragma unroll
    for (int d = 0; d < ATT; ++d) s += tanhf(s_ho[d] + hj[d]) * s_phi[d];
    s_s[j] = s;
    lmax = fmaxf(lmax, s);
  }
  float wm = warp_max32(lmax);
  if (lane == 0) s_red[wid] = wm;
  __syncthreads();
  float mx = s_red[0];
  for (int w = 1; w < 8; ++w) mx = fmaxf(mx, s_red[w]);
  __syncthreads();

  // pass 2: exp + sum
  float lsum = 0.0f;
  for (int j = tid; j < NN; j += 256) {
    const float e = expf(s_s[j] - mx);
    s_s[j] = e;
    lsum += e;
  }
  float wsum = warp_sum32(lsum);
  if (lane == 0) s_red[wid] = wsum;
  __syncthreads();
  float tot = 0.0f;
  for (int w = 0; w < 8; ++w) tot += s_red[w];
  const float rcp = 1.0f / tot;

  // pass 3: alpha store (f16) + ctx accumulation
  float cacc[32];
#pragma unroll
  for (int d = 0; d < ATT; ++d) cacc[d] = 0.0f;
  _Float16* arow = alpha + (size_t)bi * NN;
  for (int j = tid; j < NN; j += 256) {
    const float aj = s_s[j] * rcp;
    arow[j] = (_Float16)aj;
    const float* hj = H1 + (size_t)(b * NN + j) * ATT;
#pragma unroll
    for (int d = 0; d < ATT; ++d) cacc[d] += aj * tanhf(s_ho[d] + hj[d]);
  }
#pragma unroll
  for (int d = 0; d < ATT; ++d) {
    const float r = warp_sum32(cacc[d]);
    if (lane == 0) s_cp[wid * ATT + d] = r;
  }
  __syncthreads();
  if (tid < ATT) {
    float a = 0.0f;
    for (int w = 0; w < 8; ++w) a += s_cp[w * ATT + tid];
    ctx[(size_t)bi * ATT + tid] = a;
  }
}

// ---------------------------------------------------------------------------
// SE block helpers + elementwise glue
// ---------------------------------------------------------------------------
__global__ void k_colsum(const float* __restrict__ X, float* __restrict__ out) {
  const int b = blockIdx.x, d = threadIdx.x;  // 128 threads
  float a = 0.0f;
  const float* p = X + (size_t)b * 512 * 128 + d;
  for (int n = 0; n < 512; ++n) a += p[(size_t)n * 128];
  out[b * 128 + d] = a;
}

__global__ void k_se(const float* __restrict__ ys, const float* __restrict__ w1,
                     const float* __restrict__ b1, const float* __restrict__ w2,
                     const float* __restrict__ b2, float* __restrict__ y2) {
  const int b = blockIdx.x, tid = threadIdx.x;  // 128 threads
  __shared__ float s_y[128], s_y1[32];
  s_y[tid] = ys[b * 128 + tid];
  __syncthreads();
  if (tid < 32) {
    float a = b1[tid];
    for (int k = 0; k < 128; ++k) a += w1[tid * 128 + k] * s_y[k];
    s_y1[tid] = fmaxf(a, 0.0f);
  }
  __syncthreads();
  float a = b2[tid];
  for (int r = 0; r < 32; ++r) a += w2[tid * 32 + r] * s_y1[r];
  y2[b * 128 + tid] = 1.0f / (1.0f + expf(-a));
}

__global__ void k_scale(const float* __restrict__ X, const float* __restrict__ y2,
                        float* __restrict__ out) {
  const int idx = blockIdx.x * blockDim.x + threadIdx.x;
  if (idx >= 4096 * 128) return;
  const int d = idx & 127;
  const int bb = idx >> 16;  // / (512*128)
  out[idx] = X[idx] * y2[bb * 128 + d];
}

__global__ void k_combine(const float* __restrict__ Xh, const float* __restrict__ T1,
                          const float* __restrict__ T2, const float* __restrict__ theta,
                          float* __restrict__ Xp) {
  const int idx = blockIdx.x * blockDim.x + threadIdx.x;
  if (idx >= 4096 * 128) return;
  const int d = idx & 127;
  Xp[idx] = Xh[idx] + tanhf(T1[idx] * theta[2 * d]) + tanhf(T2[idx] * theta[2 * d + 1]);
}

// ---------------------------------------------------------------------------
extern "C" void kernel_launch(void* const* d_in, const int* in_sizes, int n_in,
                              void* d_out, int out_size, void* d_ws, size_t ws_size,
                              hipStream_t stream) {
  (void)in_sizes; (void)n_in; (void)out_size; (void)ws_size;
  const float* X        = (const float*)d_in[0];
  const float* A        = (const float*)d_in[1];
  const float* Ac       = (const float*)d_in[2];
  const float* embed_w  = (const float*)d_in[3];
  const float* embed_b  = (const float*)d_in[4];
  const float* topo_eps = (const float*)d_in[5];
  const float* topo_w   = (const float*)d_in[6];
  const float* topo_b   = (const float*)d_in[7];
  const float* theta    = (const float*)d_in[8];
  const float* gin_eps  = (const float*)d_in[9];
  const float* gin_w    = (const float*)d_in[10];
  const float* gin_b    = (const float*)d_in[11];
  const float* wo       = (const float*)d_in[12];
  const float* w1       = (const float*)d_in[13];
  const float* wphi     = (const float*)d_in[14];
  const float* wy_w     = (const float*)d_in[15];
  const float* wy_b     = (const float*)d_in[16];
  const float* we_w     = (const float*)d_in[17];
  const float* we_b     = (const float*)d_in[18];
  const float* l1w      = (const float*)d_in[19];
  const float* l1b      = (const float*)d_in[20];
  const float* l2w      = (const float*)d_in[21];
  const float* l2b      = (const float*)d_in[22];
  const float* mlp_w    = (const float*)d_in[23];
  const float* mlp_b    = (const float*)d_in[24];
  const float* h1w      = (const float*)d_in[25];
  const float* h1b      = (const float*)d_in[26];
  const float* h2w      = (const float*)d_in[27];
  const float* h2b      = (const float*)d_in[28];
  float* out = (float*)d_out;

  float* ws = (float*)d_ws;
  const size_t BNH = (size_t)4096 * 128;   // 524288
  float* Xh  = ws + 0 * BNH;
  float* Xp  = ws + 1 * BNH;
  float* S1  = ws + 2 * BNH;
  float* T1  = ws + 3 * BNH;
  float* T2  = ws + 4 * BNH;
  float* Xy  = ws + 5 * BNH;
  float* HOb = ws + 6 * BNH;                       // 4096*32
  float* H1b = HOb + (size_t)4096 * 32;
  float* CTX = H1b + (size_t)4096 * 32;
  float* YS  = CTX + (size_t)4096 * 32;            // 8*128
  float* Y2  = YS + 8 * 128;
  float* HH  = Y2 + 8 * 128;                       // 4096*64
  float* fend = HH + (size_t)4096 * 64;
  _Float16* A16   = (_Float16*)fend;               // 8*512*512 halves
  _Float16* Ac16  = A16 + (size_t)8 * 512 * 512;
  _Float16* ALPHA = Ac16 + (size_t)8 * 512 * 512;  // 8*512*512 halves
  _Float16* T16   = ALPHA + (size_t)8 * 512 * 512; // 8*128*512 halves

  auto lin = [&](const float* Xi, const float* Wi, const float* bi,
                 const float* a0, const float* a1, float* o,
                 int M, int K, int Nc, int relu) {
    if (Nc >= 64) {
      const int tiles = (M / 16) * ((Nc + 63) / 64);
      k_linear_t<4><<<(tiles + 7) / 8, 256, 0, stream>>>(Xi, Wi, bi, a0, a1, o,
                                                         M, K, Nc, relu);
    } else if (Nc >= 32) {
      const int tiles = (M / 16) * ((Nc + 31) / 32);
      k_linear_t<2><<<(tiles + 7) / 8, 256, 0, stream>>>(Xi, Wi, bi, a0, a1, o,
                                                         M, K, Nc, relu);
    } else {
      const int tiles = (M / 16) * ((Nc + 15) / 16);
      k_linear_t<1><<<(tiles + 7) / 8, 256, 0, stream>>>(Xi, Wi, bi, a0, a1, o,
                                                         M, K, Nc, relu);
    }
  };
  auto bmm = [&](const _Float16* P16, const _Float16* XTi, const float* Xadd,
                 const float* ep, float* o) {
    const int tiles = 8 * (512 / 16) * (128 / 64);  // 512 waves
    k_bmm16<<<(tiles + 7) / 8, 256, 0, stream>>>(P16, XTi, Xadd, ep, o, 8, 512, 128);
  };
  auto transpose = [&](const float* Xi, _Float16* XTo) {
    k_transpose16<<<dim3(512 / 32, 128 / 32, 8), 256, 0, stream>>>(Xi, XTo, 512, 128);
  };

  // --- one-time f16 adjacency conversion (exact: values are 0/1) ---
  const int ADJ = 8 * 512 * 512;
  k_cvt_f16<<<(ADJ + 255) / 256, 256, 0, stream>>>(A, A16, ADJ);
  k_cvt_f16<<<(ADJ + 255) / 256, 256, 0, stream>>>(Ac, Ac16, ADJ);

  // --- topo positional embedding ---
  lin(X, embed_w, embed_b, nullptr, nullptr, Xh, 4096, 64, 128, 0);
  transpose(Xh, T16);
  bmm(A16, T16, Xh, topo_eps, S1);
  lin(S1, topo_w, topo_b, nullptr, nullptr, T1, 4096, 128, 128, 1);
  bmm(Ac16, T16, Xh, topo_eps, S1);
  lin(S1, topo_w, topo_b, nullptr, nullptr, T2, 4096, 128, 128, 1);
  k_combine<<<2048, 256, 0, stream>>>(Xh, T1, T2, theta, Xp);

  // --- layers ---
  for (int l = 0; l < 3; ++l) {
    // dual GIN: T2 = relu(GIN_A) + relu(GIN_Ac)
    transpose(Xp, T16);
    bmm(A16, T16, Xp, gin_eps + 2 * l + 0, S1);
    lin(S1, gin_w + (size_t)(2 * l + 0) * 128 * 128, gin_b + (2 * l + 0) * 128,
        nullptr, nullptr, T1, 4096, 128, 128, 1);
    bmm(Ac16, T16, Xp, gin_eps + 2 * l + 1, S1);
    lin(S1, gin_w + (size_t)(2 * l + 1) * 128 * 128, gin_b + (2 * l + 1) * 128,
        T1, nullptr, T2, 4096, 128, 128, 1);
    // attention
    lin(Xp, wo + (size_t)l * 32 * 128, nullptr, nullptr, nullptr, HOb, 4096, 128, 32, 0);
    lin(Xp, w1 + (size_t)l * 32 * 128, nullptr, nullptr, nullptr, H1b, 4096, 128, 32, 0);
    k_attn<<<4096, 256, 0, stream>>>(HOb, H1b, wphi + l * 32, ALPHA, CTX);
    lin(Xp, wy_w + (size_t)l * 128 * 128, wy_b + l * 128, nullptr, nullptr, Xy,
        4096, 128, 128, 0);
    transpose(Xy, T16);
    bmm(ALPHA, T16, nullptr, nullptr, S1);  // S1 = alpha @ Xy
    // T1 = Xbar = ctx@we^T + we_b + (alpha@Xy) + (XA+XAC)
    lin(CTX, we_w + (size_t)l * 128 * 32, we_b + l * 128, S1, T2, T1, 4096, 32, 128, 0);
    // SE gate
    k_colsum<<<8, 128, 0, stream>>>(T1, YS);
    k_se<<<8, 128, 0, stream>>>(YS, l1w + l * 32 * 128, l1b + l * 32,
                                l2w + l * 128 * 32, l2b + l * 128, Y2);
    k_scale<<<2048, 256, 0, stream>>>(T1, Y2, S1);
    lin(S1, mlp_w + (size_t)l * 128 * 128, mlp_b + l * 128, nullptr, nullptr, Xp,
        4096, 128, 128, 0);
  }

  // --- node head ---
  lin(Xp, h1w, h1b, nullptr, nullptr, HH, 4096, 128, 64, 1);
  lin(HH, h2w, h2b, nullptr, nullptr, out, 4096, 64, 3, 0);
}
